// TreeLSTM_2860448219907
// MI455X (gfx1250) — matrix operand order repaired and checked by reading, not compiled, not run on hardware
//
#include <hip/hip_runtime.h>
#include <hip/hip_bf16.h>

// ---------------- types ----------------
typedef __attribute__((ext_vector_type(16))) __bf16 v16bf;
typedef __attribute__((ext_vector_type(8)))  __bf16 v8bf;
typedef __attribute__((ext_vector_type(8)))  float  v8f;

union V16U { v16bf v; struct { v8bf lo, hi; } s; };

__device__ __forceinline__ __bf16 f2bf(float f) {
    union { float f; unsigned u; } x; x.f = f;
    unsigned r = (x.u + 0x7FFFu + ((x.u >> 16) & 1u)) >> 16;   // RNE
    union { unsigned short s; __bf16 b; } y; y.s = (unsigned short)r;
    return y.b;
}
__device__ __forceinline__ float sigm(float x) { return 1.f / (1.f + __expf(-x)); }

__device__ __forceinline__ v8f wmma_bf16(v16bf a, v16bf b, v8f c) {
    return __builtin_amdgcn_wmma_f32_16x16x32_bf16(false, a, false, b, (short)0, c,
                                                   false, false);
}

// A fragment from a bf16 row (row-major, per-lane row pointer).
// ISA 7.12.2 (16-bit A 16x32): lane half h = lane>>4; elements 0..7 hold
// K = k0 + h*8 + [0,8); elements 8..15 hold K = k0 + 16 + h*8 + [0,8).
// -> two contiguous 16-byte loads.
__device__ __forceinline__ v16bf load_a_frag_bf(const __bf16* rowptr, int lane, int k0) {
    const __bf16* p = rowptr + k0 + ((lane >> 4) << 3);
    V16U u;
    u.s.lo = *reinterpret_cast<const v8bf*>(p);
    u.s.hi = *reinterpret_cast<const v8bf*>(p + 16);
    return u.v;
}

// B fragment: 32x16 bf16 tile from row-major bf16 weight G[N][K] (ldk elems).
// lane l holds weight row n0+(l&15), K = k0 + (l>>4)*8 + e (e<8), k0 + 16 + (l>>4)*8 + (e-8) (e>=8).
__device__ __forceinline__ v16bf load_b_frag(const __bf16* G, int ldk, int n0, int k0,
                                             int lane) {
    const __bf16* p = G + (size_t)(n0 + (lane & 15)) * ldk + k0 + ((lane >> 4) << 3);
    V16U u;
    u.s.lo = *reinterpret_cast<const v8bf*>(p);
    u.s.hi = *reinterpret_cast<const v8bf*>(p + 16);
    return u.v;
}

// ---------------- problem constants ----------------
#define Bsz 128
#define Nn  1023
#define Hh  256
#define Xx  256
#define Cc  10
#define Vv  32000

// ---------------- workspace layout (bytes) ----------------
static constexpr size_t OFF_WBF = 0;                                 // 768*256 bf16
static constexpr size_t OFF_GBF = OFF_WBF + (size_t)768 * 256 * 2;   // 1280*512 bf16
static constexpr size_t OFF_MT  = OFF_GBF + (size_t)1280 * 512 * 2;  // 256*256 bf16
static constexpr size_t OFF_EMB = OFF_MT + (size_t)256 * 256 * 2;    // V*X bf16
static constexpr size_t OFF_H   = OFF_EMB + (size_t)Vv * Xx * 2;     // f32 h
static constexpr size_t SZ_HC   = (size_t)Bsz * Nn * Hh * 4;
static constexpr size_t OFF_C   = OFF_H + SZ_HC;                     // f32 c
static constexpr size_t OFF_HBF = OFF_C + SZ_HC;                     // bf16 shadow of h
static constexpr size_t OFF_Q   = OFF_HBF + (size_t)Bsz * Nn * Hh * 2;
static constexpr size_t OFF_Y   = OFF_Q + (size_t)Bsz * Hh * 4;      // level staging

// ---------------- kernels ----------------

// Convert weights + embedding table to bf16.
// Wbf = W_iou; Gbf = [Uf_W ; U_iou]; Mt = Wmem^T; embbf = emb.
__global__ void k_convert(const float* W_iou, const float* U_iou, const float* Uf_W,
                          const float* Wmem, const float* emb, __bf16* Wbf,
                          __bf16* Gbf, __bf16* Mt, __bf16* embbf) {
    int i = blockIdx.x * blockDim.x + threadIdx.x;
    const int nW = 768 * 256, nG = 1280 * 512, nM = 256 * 256, nE = Vv * Xx;
    if (i < nW) Wbf[i] = f2bf(W_iou[i]);
    if (i < nG) {
        int row = i >> 9, k = i & 511;
        float v = (row < 512) ? Uf_W[row * 512 + k] : U_iou[(row - 512) * 512 + k];
        Gbf[i] = f2bf(v);
    }
    if (i < nM) {
        int h = i >> 8, g = i & 255;
        Mt[i] = f2bf(Wmem[g * 256 + h]);   // transpose so q = dec @ Mt^T
    }
    if (i < nE) embbf[i] = f2bf(emb[i]);
}

// Leaf level: iou = emb[word] @ W_iou^T (M=65536,N=768,K=256), fused node epilogue.
// Block = 8 waves = 16 output rows x 768 cols (wave owns 6 16x16 tiles).
__global__ void k_leaf(const int* wordid, const __bf16* embbf, const __bf16* Wbf,
                       const float* b_iou, float* hbuf, float* cbuf, __bf16* hbf) {
    __shared__ float sY[16][768];
    int tid = threadIdx.x, lane = tid & 31, wave = tid >> 5;
    int blockRow = blockIdx.x << 4;
    int r = blockRow + (lane & 15);
    int b = r >> 9;
    int node = 511 + (r & 511);
    int word = wordid[b * Nn + node];
    const __bf16* arow = embbf + (size_t)word * Xx;

    v8f acc[6] = {};
    for (int k0 = 0; k0 < Xx; k0 += 32) {
        v16bf a = load_a_frag_bf(arow, lane, k0);
#pragma unroll
        for (int t = 0; t < 6; ++t)
            acc[t] = wmma_bf16(a, load_b_frag(Wbf, Xx, (wave * 6 + t) << 4, k0, lane),
                               acc[t]);
    }
    int mrow = (lane >> 4) << 3;
#pragma unroll
    for (int t = 0; t < 6; ++t) {
        int ncol = ((wave * 6 + t) << 4) + (lane & 15);
#pragma unroll
        for (int rr = 0; rr < 8; ++rr) sY[mrow + rr][ncol] = acc[t][rr];
    }
    __syncthreads();

    for (int i = tid; i < 16 * Hh; i += 256) {
        int row = i >> 8, hc = i & 255;
        int gr = blockRow + row;
        int b2 = gr >> 9, node2 = 511 + (gr & 511);
        float iv = sY[row][hc] + b_iou[hc];
        float ov = sY[row][256 + hc] + b_iou[256 + hc];
        float uv = sY[row][512 + hc] + b_iou[512 + hc];
        float cn = sigm(iv) * tanhf(uv);         // c_in = c0 = 0
        float hn = sigm(ov) * tanhf(cn);
        size_t off = ((size_t)b2 * Nn + node2) * Hh + hc;
        hbuf[off] = hn;
        cbuf[off] = cn;
        hbf[off]  = f2bf(hn);
    }
}

// Interior level GEMM: Y[M x 1280] = h_cat @ [Uf_W;U_iou]^T, M = 128*npar, K = 512.
// Block = 8 waves = 16 rows x 1280 cols (wave owns 10 tiles). A from bf16 h shadow.
__global__ void k_level_gemm(const __bf16* hbf, const __bf16* Gbf, float* Ybuf,
                             int ps, int lg) {
    int tid = threadIdx.x, lane = tid & 31, wave = tid >> 5;
    int blockRow = blockIdx.x << 4;
    int rr = blockRow + (lane & 15);
    int b = rr >> lg, p = rr & ((1 << lg) - 1);
    int q = ps + p;
    const __bf16* arow = hbf + ((size_t)b * Nn + 2 * q + 1) * Hh;  // children contiguous

    v8f acc[10] = {};
    for (int k0 = 0; k0 < 512; k0 += 32) {
        v16bf a = load_a_frag_bf(arow, lane, k0);
#pragma unroll
        for (int t = 0; t < 10; ++t)
            acc[t] = wmma_bf16(a, load_b_frag(Gbf, 512, (wave * 10 + t) << 4, k0, lane),
                               acc[t]);
    }
    int mrow = (lane >> 4) << 3;
#pragma unroll
    for (int t = 0; t < 10; ++t) {
        int ncol = ((wave * 10 + t) << 4) + (lane & 15);
#pragma unroll
        for (int r = 0; r < 8; ++r)
            Ybuf[(size_t)(blockRow + mrow + r) * 1280 + ncol] = acc[t][r];
    }
}

// Interior level pointwise node: f-gates, c_sum, iou -> (h,c) of parent.
__global__ void k_level_node(const float* Y, const float* b_iou, const float* Uf_b,
                             float* hbuf, float* cbuf, __bf16* hbf, int ps, int lg) {
    int rr = blockIdx.x;
    int hc = threadIdx.x;
    int b = rr >> lg, p = rr & ((1 << lg) - 1);
    int q = ps + p;
    const float* yr = Y + (size_t)rr * 1280;
    size_t base = (size_t)b * Nn;
    float fl = sigm(yr[hc] + Uf_b[hc]);
    float fr = sigm(yr[256 + hc] + Uf_b[256 + hc]);
    float csum = fl * cbuf[(base + 2 * q + 1) * Hh + hc]
               + fr * cbuf[(base + 2 * q + 2) * Hh + hc];
    float iv = yr[512 + hc] + b_iou[hc];
    float ov = yr[768 + hc] + b_iou[256 + hc];
    float uv = yr[1024 + hc] + b_iou[512 + hc];
    float cn = sigm(iv) * tanhf(uv) + csum;
    float hn = sigm(ov) * tanhf(cn);
    size_t off = (base + q) * Hh + hc;
    hbuf[off] = hn;
    cbuf[off] = cn;
    hbf[off]  = f2bf(hn);
}

// q[b] = Wmem^T @ dec_b  (GEMM M=128,N=256,K=256 vs transposed Wmem in bf16).
__global__ void k_qproj(const __bf16* hbf, const __bf16* Mt, float* qbuf) {
    int tid = threadIdx.x, lane = tid & 31, wave = tid >> 5;
    int blockRow = blockIdx.x << 4;
    int b = blockRow + (lane & 15);
    const __bf16* arow = hbf + (size_t)b * Nn * Hh;   // h[b][0][:]

    v8f acc[2] = {};
    for (int k0 = 0; k0 < Hh; k0 += 32) {
        v16bf a = load_a_frag_bf(arow, lane, k0);
#pragma unroll
        for (int t = 0; t < 2; ++t)
            acc[t] = wmma_bf16(a, load_b_frag(Mt, Hh, (wave * 2 + t) << 4, k0, lane),
                               acc[t]);
    }
    int mrow = (lane >> 4) << 3;
#pragma unroll
    for (int t = 0; t < 2; ++t) {
        int ncol = ((wave * 2 + t) << 4) + (lane & 15);
#pragma unroll
        for (int r = 0; r < 8; ++r)
            qbuf[(size_t)(blockRow + mrow + r) * Hh + ncol] = acc[t][r];
    }
}

// Attention + head: scores = q.h, softmax, context, relu-linear, linear. 1 block/batch.
__global__ void k_attn(const float* hbuf, const float* qbuf, const float* wh_W,
                       const float* wh_b, const float* lin_W, const float* lin_b,
                       float* out) {
    __shared__ float sq[Hh];
    __shared__ float sw[Nn + 1];
    __shared__ float red[256];
    __shared__ float sctx[Hh];
    int b = blockIdx.x, tid = threadIdx.x;
    sq[tid] = qbuf[b * Hh + tid];
    __syncthreads();

    const float* hb = hbuf + (size_t)b * Nn * Hh;
    float lmax = -3.4e38f;
    for (int n = tid; n < Nn; n += 256) {
        const float4* hr = reinterpret_cast<const float4*>(hb + (size_t)n * Hh);
        float s = 0.f;
#pragma unroll 8
        for (int k = 0; k < Hh / 4; ++k) {
            float4 v = hr[k];
            s += v.x * sq[4 * k] + v.y * sq[4 * k + 1] + v.z * sq[4 * k + 2] +
                 v.w * sq[4 * k + 3];
        }
        sw[n] = s;
        lmax = fmaxf(lmax, s);
    }
    red[tid] = lmax;
    __syncthreads();
    for (int st = 128; st > 0; st >>= 1) {
        if (tid < st) red[tid] = fmaxf(red[tid], red[tid + st]);
        __syncthreads();
    }
    float mx = red[0];
    __syncthreads();

    float lsum = 0.f;
    for (int n = tid; n < Nn; n += 256) {
        float w = __expf(sw[n] - mx);
        sw[n] = w;
        lsum += w;
    }
    red[tid] = lsum;
    __syncthreads();
    for (int st = 128; st > 0; st >>= 1) {
        if (tid < st) red[tid] += red[tid + st];
        __syncthreads();
    }
    float inv = 1.f / red[0];
    __syncthreads();

    float ctx = 0.f;
    for (int n = 0; n < Nn; ++n) {
        if (n + 8 < Nn) __builtin_prefetch(hb + (size_t)(n + 8) * Hh + tid, 0, 1);
        ctx += sw[n] * hb[(size_t)n * Hh + tid];
    }
    sctx[tid] = ctx * inv;
    __syncthreads();

    float a = wh_b[tid];
    const float* wr = wh_W + tid * Hh;
#pragma unroll 8
    for (int k = 0; k < Hh; ++k) a += wr[k] * sctx[k];
    red[tid] = fmaxf(a, 0.f);
    __syncthreads();

    if (tid < Cc) {
        float o = lin_b[tid];
        const float* lr = lin_W + tid * Hh;
#pragma unroll 8
        for (int k = 0; k < Hh; ++k) o += lr[k] * red[k];
        out[b * Cc + tid] = o;
    }
}

// ---------------- host launcher ----------------
extern "C" void kernel_launch(void* const* d_in, const int* in_sizes, int n_in,
                              void* d_out, int out_size, void* d_ws, size_t ws_size,
                              hipStream_t stream) {
    const int*   wordid = (const int*)d_in[0];
    // d_in[1]=h0, d_in[2]=c0 are zeros -> folded analytically
    const float* emb    = (const float*)d_in[3];
    const float* W_iou  = (const float*)d_in[4];
    const float* U_iou  = (const float*)d_in[5];
    const float* b_iou  = (const float*)d_in[6];
    const float* Uf_W   = (const float*)d_in[7];
    const float* Uf_b   = (const float*)d_in[8];
    const float* Wmem   = (const float*)d_in[9];
    const float* wh_W   = (const float*)d_in[10];
    const float* wh_b   = (const float*)d_in[11];
    const float* lin_W  = (const float*)d_in[12];
    const float* lin_b  = (const float*)d_in[13];
    float* out = (float*)d_out;

    char* ws = (char*)d_ws;
    __bf16* Wbf   = (__bf16*)(ws + OFF_WBF);
    __bf16* Gbf   = (__bf16*)(ws + OFF_GBF);
    __bf16* Mt    = (__bf16*)(ws + OFF_MT);
    __bf16* embbf = (__bf16*)(ws + OFF_EMB);
    float*  hbuf  = (float*)(ws + OFF_H);
    float*  cbuf  = (float*)(ws + OFF_C);
    __bf16* hbf   = (__bf16*)(ws + OFF_HBF);
    float*  qbuf  = (float*)(ws + OFF_Q);
    float*  Ybuf  = (float*)(ws + OFF_Y);

    // covers the largest conversion range (emb: V*X elements)
    k_convert<<<dim3((Vv * Xx + 255) / 256), dim3(256), 0, stream>>>(
        W_iou, U_iou, Uf_W, Wmem, emb, Wbf, Gbf, Mt, embbf);

    // leaves: 65536 rows / 16 per block
    k_leaf<<<dim3(4096), dim3(256), 0, stream>>>(wordid, embbf, Wbf, b_iou, hbuf, cbuf,
                                                 hbf);

    // interior levels, bottom-up
    for (int level = 8; level >= 0; --level) {
        int npar = 1 << level;
        int ps   = npar - 1;
        k_level_gemm<<<dim3(8 * npar), dim3(256), 0, stream>>>(hbf, Gbf, Ybuf, ps,
                                                               level);
        k_level_node<<<dim3(Bsz * npar), dim3(256), 0, stream>>>(Ybuf, b_iou, Uf_b,
                                                                 hbuf, cbuf, hbf, ps,
                                                                 level);
    }

    k_qproj<<<dim3(8), dim3(256), 0, stream>>>(hbf, Mt, qbuf);
    k_attn<<<dim3(Bsz), dim3(256), 0, stream>>>(hbuf, qbuf, wh_W, wh_b, lin_W, lin_b,
                                                out);
}
